// devign_11879879544343
// MI455X (gfx1250) — compile-verified, matrix operand order checked by reading, not run on hardware
//
#include <hip/hip_runtime.h>
#include <hip/hip_bf16.h>
#include <cstdint>
#include <cstddef>

// ---------------------------------------------------------------------------
// Devign GGNN forward for MI455X (gfx1250, wave32, WMMA).
//  * Per-relation linear pulled out of the edge loop (503 GFLOP -> ~40 GFLOP
//    of bf16 WMMA + ~16 GB gather/scatter, the true 23.3 TB/s roofline term).
//  * rel_w [4,128,128] treated as one [512,128] weight -> ONE N x 512 WMMA
//    GEMM per step feeds all 4 relations.
//  * GEMM register-blocks 4 N-tiles per wave: A fragment loaded once per
//    K-chunk and reused by 4 v_wmma_f32_16x16x32_bf16 issues.
// ---------------------------------------------------------------------------

#define N_NODES  20000
#define W2V      100
#define TDIM     32
#define HDIM     128
#define H3       (3 * HDIM)
#define H4       (4 * HDIM)
#define T_STEPS  6
#define RELN     4
#define L_IN     20000
#define L_P1     10000
#define L_P2     5000
#define FLATSZ   (64 * L_P2)

typedef __bf16 bf16_t;
typedef __attribute__((ext_vector_type(16))) __bf16 v16bf;
typedef __attribute__((ext_vector_type(8)))  float  v8f;

union FragBF {
    v16bf v;
    uint4 q[2];
};

static __host__ __device__ inline unsigned cdiv_u(long a, long b) {
    return (unsigned)((a + b - 1) / b);
}

// ---------------------------------------------------------------------------
// Utility kernels
// ---------------------------------------------------------------------------
__global__ void k_zero_f32(float* __restrict__ p, long n) {
    long i = (long)blockIdx.x * blockDim.x + threadIdx.x;
    if (i < n) p[i] = 0.0f;
}

__global__ void k_f32_to_bf16(const float* __restrict__ s, bf16_t* __restrict__ d, long n) {
    long i = (long)blockIdx.x * blockDim.x + threadIdx.x;
    if (i < n) d[i] = (bf16_t)s[i];
}

// ---------------------------------------------------------------------------
// Input projection: x0 = [x_code | type_emb[type_ids]] @ in_proj_w^T + b
// Also seeds h (f32) and h_bf16. Tiny (0.67 GFLOP) -> plain VALU.
// ---------------------------------------------------------------------------
__global__ __launch_bounds__(256)
void k_in_proj(const float* __restrict__ x_code, const int* __restrict__ type_ids,
               const float* __restrict__ type_emb,
               const float* __restrict__ W, const float* __restrict__ b,
               float* __restrict__ x0, float* __restrict__ h, bf16_t* __restrict__ hb) {
    long t = (long)blockIdx.x * blockDim.x + threadIdx.x;
    if (t >= (long)N_NODES * HDIM) return;
    int n = (int)(t >> 7);
    int j = (int)(t & 127);
    const float* w  = W + (size_t)j * (W2V + TDIM);
    const float* xc = x_code + (size_t)n * W2V;
    const float* te = type_emb + (size_t)type_ids[n] * TDIM;
    float acc = b[j];
#pragma unroll 4
    for (int k = 0; k < W2V; ++k) acc += xc[k] * w[k];
#pragma unroll 4
    for (int k = 0; k < TDIM; ++k) acc += te[k] * w[W2V + k];
    x0[t] = acc;
    h[t]  = acc;
    hb[t] = (bf16_t)acc;
}

// ---------------------------------------------------------------------------
// WMMA GEMM, 4-wide N register blocking:
//   C[M,Nout] f32 = A[M,K]bf16 @ (W[Nout,K]bf16)^T + bias
// One wave -> one 16x64 output strip (4 accumulators). Per 32-K chunk the A
// fragment is loaded once (two 16B loads/lane) and reused by 4 WMMA issues.
// Fragment layout (16-bit operands, wave32): lane&15 = row in tile,
// lane>>4 = K-half; element slices at k and k+16. B = W^T has N striped
// across lanes, so the B fragment is the same pattern applied to W's rows.
// Requires: M%16==0, Nout%64==0, K%32==0 (true for all call sites).
// ---------------------------------------------------------------------------
__global__ __launch_bounds__(256)
void k_wmma_gemm_bf16(const bf16_t* __restrict__ A, const bf16_t* __restrict__ W,
                      const float* __restrict__ bias, float* __restrict__ C,
                      int M, int Nout, int K) {
    const int lane   = threadIdx.x & 31;
    const int wave   = threadIdx.x >> 5;
    const int nstrip = Nout >> 6;                // 64-wide N strips
    const int mtiles = M >> 4;
    int strip = blockIdx.x * 8 + wave;
    if (strip >= mtiles * nstrip) return;        // wave-uniform: EXEC stays all-1s
    const int mt  = strip / nstrip;
    const int ns  = strip % nstrip;
    const int half = lane >> 4;                  // which 8-wide K slice this lane owns
    const int idx  = lane & 15;                  // row within a 16-row tile

    const bf16_t* arow = A + (size_t)(mt * 16 + idx) * K + half * 8;
    const bf16_t* wrow = W + (size_t)(ns * 64 + idx) * K + half * 8;
    const size_t  wstep = (size_t)16 * K;        // next 16-row W sub-tile

    v8f acc0 = {}, acc1 = {}, acc2 = {}, acc3 = {};
#pragma unroll
    for (int k0 = 0; k0 < 128; k0 += 32) {       // K == 128 at every call site
        FragBF a;
        a.q[0] = *(const uint4*)(arow + k0);
        a.q[1] = *(const uint4*)(arow + k0 + 16);
        FragBF b0, b1, b2, b3;
        b0.q[0] = *(const uint4*)(wrow + k0);
        b0.q[1] = *(const uint4*)(wrow + k0 + 16);
        b1.q[0] = *(const uint4*)(wrow + wstep + k0);
        b1.q[1] = *(const uint4*)(wrow + wstep + k0 + 16);
        b2.q[0] = *(const uint4*)(wrow + 2 * wstep + k0);
        b2.q[1] = *(const uint4*)(wrow + 2 * wstep + k0 + 16);
        b3.q[0] = *(const uint4*)(wrow + 3 * wstep + k0);
        b3.q[1] = *(const uint4*)(wrow + 3 * wstep + k0 + 16);
        acc0 = __builtin_amdgcn_wmma_f32_16x16x32_bf16(false, a.v, false, b0.v, (short)0, acc0, false, false);
        acc1 = __builtin_amdgcn_wmma_f32_16x16x32_bf16(false, a.v, false, b1.v, (short)0, acc1, false, false);
        acc2 = __builtin_amdgcn_wmma_f32_16x16x32_bf16(false, a.v, false, b2.v, (short)0, acc2, false, false);
        acc3 = __builtin_amdgcn_wmma_f32_16x16x32_bf16(false, a.v, false, b3.v, (short)0, acc3, false, false);
    }

    const int row0 = mt * 16 + half * 8;         // C VGPR j -> row0 + j
    v8f accs[4] = { acc0, acc1, acc2, acc3 };
#pragma unroll
    for (int b = 0; b < 4; ++b) {
        const int col = ns * 64 + b * 16 + idx;
        const float bv = bias ? bias[col] : 0.0f;
#pragma unroll
        for (int j = 0; j < 8; ++j)
            C[(size_t)(row0 + j) * Nout + col] = accs[b][j] + bv;
    }
}

// ---------------------------------------------------------------------------
// Edge scatter: m[dst] += hw[src] + rel_b   (32 lanes/edge, float4 each,
// global_atomic_add_f32). hw rows are RELN*HDIM wide (fused-relation GEMM);
// the caller passes hw already offset to its relation's 128-column slice.
// ---------------------------------------------------------------------------
__global__ __launch_bounds__(256)
void k_scatter_add(const int* __restrict__ src, const int* __restrict__ dst,
                   const float* __restrict__ hw, const float* __restrict__ relb,
                   float* __restrict__ m, int E) {
    long t = (long)blockIdx.x * blockDim.x + threadIdx.x;
    int e  = (int)(t >> 5);
    if (e >= E) return;
    int c4 = (int)(t & 31) * 4;
    int s  = src[e];
    int d  = dst[e];
    const float4 v = *(const float4*)(hw + (size_t)s * H4 + c4);
    const float4 b = *(const float4*)(relb + c4);
    float* p = m + (size_t)d * HDIM + c4;
    atomicAdd(p + 0, v.x + b.x);
    atomicAdd(p + 1, v.y + b.y);
    atomicAdd(p + 2, v.z + b.z);
    atomicAdd(p + 3, v.w + b.w);
}

// ---------------------------------------------------------------------------
// Fused GRU gates: consumes gi = m@Wih^T+bih, gh = h@Whh^T+bhh; rewrites h and
// its bf16 mirror for the next step's WMMA GEMMs.
// ---------------------------------------------------------------------------
__global__ __launch_bounds__(256)
void k_gru_gate(const float* __restrict__ gi, const float* __restrict__ gh,
                float* __restrict__ h, bf16_t* __restrict__ hb) {
    long t = (long)blockIdx.x * blockDim.x + threadIdx.x;
    if (t >= (long)N_NODES * HDIM) return;
    int n = (int)(t >> 7);
    int c = (int)(t & 127);
    const float* gin = gi + (size_t)n * H3;
    const float* ghn = gh + (size_t)n * H3;
    float r  = 1.0f / (1.0f + __expf(-(gin[c] + ghn[c])));
    float z  = 1.0f / (1.0f + __expf(-(gin[HDIM + c] + ghn[HDIM + c])));
    float ng = tanhf(gin[2 * HDIM + c] + r * ghn[2 * HDIM + c]);
    float hn = (1.0f - z) * ng + z * h[t];
    h[t]  = hn;
    hb[t] = (bf16_t)hn;
}

// ---------------------------------------------------------------------------
// Conv head (total ~2 GFLOP; simple VALU kernels).
// channels: ci<128 -> x0, ci>=128 -> h;  w1[o][ci][dk], K=3, pad=1, ReLU.
// ---------------------------------------------------------------------------
__global__ __launch_bounds__(256)
void k_conv1(const float* __restrict__ x0, const float* __restrict__ h,
             const float* __restrict__ w, const float* __restrict__ b,
             float* __restrict__ out) {
    long t = (long)blockIdx.x * blockDim.x + threadIdx.x;
    if (t >= (long)64 * L_IN) return;
    int o = (int)(t / L_IN);
    int n = (int)(t % L_IN);
    const float* wo = w + (size_t)o * 256 * 3;
    float acc = b[o];
#pragma unroll
    for (int dk = 0; dk < 3; ++dk) {
        int nn = n + dk - 1;
        if (nn < 0 || nn >= L_IN) continue;
        const float* xr = x0 + (size_t)nn * HDIM;
        const float* hr = h  + (size_t)nn * HDIM;
#pragma unroll 4
        for (int ci = 0; ci < HDIM; ++ci) acc += xr[ci] * wo[ci * 3 + dk];
#pragma unroll 4
        for (int ci = 0; ci < HDIM; ++ci) acc += hr[ci] * wo[(HDIM + ci) * 3 + dk];
    }
    out[t] = fmaxf(acc, 0.0f);
}

__global__ __launch_bounds__(256)
void k_pool1(const float* __restrict__ in, float* __restrict__ out) {
    long t = (long)blockIdx.x * blockDim.x + threadIdx.x;   // 64 * L_P1
    if (t >= (long)64 * L_P1) return;
    int o = (int)(t / L_P1);
    int i = (int)(t % L_P1);
    float m = -INFINITY;
#pragma unroll
    for (int d = -1; d <= 1; ++d) {
        int nn = 2 * i + d;
        if (nn >= 0 && nn < L_IN) m = fmaxf(m, in[(size_t)o * L_IN + nn]);
    }
    out[t] = m;
}

__global__ __launch_bounds__(256)
void k_conv2(const float* __restrict__ in, const float* __restrict__ w,
             const float* __restrict__ b, float* __restrict__ out) {
    long t = (long)blockIdx.x * blockDim.x + threadIdx.x;   // 64 * L_P1
    if (t >= (long)64 * L_P1) return;
    int o = (int)(t / L_P1);
    int i = (int)(t % L_P1);
    const float* wo = w + (size_t)o * 64;
    float acc = b[o];
#pragma unroll 4
    for (int ci = 0; ci < 64; ++ci) acc += wo[ci] * in[(size_t)ci * L_P1 + i];
    out[t] = fmaxf(acc, 0.0f);
}

__global__ __launch_bounds__(256)
void k_pool2(const float* __restrict__ in, float* __restrict__ out) {
    long t = (long)blockIdx.x * blockDim.x + threadIdx.x;   // 64 * L_P2
    if (t >= (long)64 * L_P2) return;
    int o = (int)(t / L_P2);
    int j = (int)(t % L_P2);
    const float* p = in + (size_t)o * L_P1 + 2 * j;
    out[t] = fmaxf(p[0], p[1]);
}

__global__ __launch_bounds__(256)
void k_fc1(const float* __restrict__ w, const float* __restrict__ b,
           const float* __restrict__ x, float* __restrict__ y) {
    __shared__ float red[256];
    int k = blockIdx.x;                               // 128 rows
    const float* wk = w + (size_t)k * FLATSZ;
    float acc = 0.0f;
    for (int f = threadIdx.x; f < FLATSZ; f += 256) acc += wk[f] * x[f];
    red[threadIdx.x] = acc;
    __syncthreads();
    for (int s = 128; s > 0; s >>= 1) {
        if ((int)threadIdx.x < s) red[threadIdx.x] += red[threadIdx.x + s];
        __syncthreads();
    }
    if (threadIdx.x == 0) y[k] = fmaxf(red[0] + b[k], 0.0f);
}

__global__ void k_fc2(const float* __restrict__ w, const float* __restrict__ b,
                      const float* __restrict__ y, float* __restrict__ out) {
    int j = threadIdx.x >> 5;                         // 2 waves, one logit each
    int lane = threadIdx.x & 31;
    if (j >= 2) return;
    float acc = 0.0f;
    for (int k = lane; k < HDIM; k += 32) acc += w[(size_t)j * HDIM + k] * y[k];
    for (int off = 16; off > 0; off >>= 1) acc += __shfl_down(acc, off, 32);
    if (lane == 0) out[j] = acc + b[j];
}

// ---------------------------------------------------------------------------
// Host-side orchestration (capture-safe: only async launches on `stream`).
// ---------------------------------------------------------------------------
static inline size_t align_up(size_t x, size_t a) { return (x + a - 1) & ~(a - 1); }

extern "C" void kernel_launch(void* const* d_in, const int* in_sizes, int n_in,
                              void* d_out, int out_size, void* d_ws, size_t ws_size,
                              hipStream_t stream) {
    (void)n_in; (void)out_size;
    const float* x_code    = (const float*)d_in[0];
    const int*   type_ids  = (const int*)d_in[1];
    const int*   edges[RELN] = { (const int*)d_in[2], (const int*)d_in[3],
                                 (const int*)d_in[4], (const int*)d_in[5] };
    const int    E         = in_sizes[2] / 2;       // edges are [2, E]
    const float* type_emb  = (const float*)d_in[6];
    const float* in_proj_w = (const float*)d_in[7];
    const float* in_proj_b = (const float*)d_in[8];
    const float* rel_w     = (const float*)d_in[9];  // [4,128,128] == [512,128]
    const float* rel_b     = (const float*)d_in[10];
    const float* gru_w_ih  = (const float*)d_in[11];
    const float* gru_w_hh  = (const float*)d_in[12];
    const float* gru_b_ih  = (const float*)d_in[13];
    const float* gru_b_hh  = (const float*)d_in[14];
    const float* conv1_w   = (const float*)d_in[15];
    const float* conv1_b   = (const float*)d_in[16];
    const float* conv2_w   = (const float*)d_in[17];
    const float* conv2_b   = (const float*)d_in[18];
    const float* fc1_w     = (const float*)d_in[19];
    const float* fc1_b     = (const float*)d_in[20];
    const float* fc2_w     = (const float*)d_in[21];
    const float* fc2_b     = (const float*)d_in[22];
    float* logits = (float*)d_out;

    // ---- workspace layout (~156 MB; assumes ws_size is sufficient) ----
    char*  base = (char*)d_ws;
    size_t off  = 0;
    auto carve = [&](size_t bytes) { off = align_up(off, 256); char* p = base + off; off += bytes; return p; };
    const size_t NH  = (size_t)N_NODES * HDIM;
    const size_t NH3 = (size_t)N_NODES * H3;
    const size_t NH4 = (size_t)N_NODES * H4;
    float*  x0    = (float*)carve(NH * 4);
    float*  h     = (float*)carve(NH * 4);
    float*  m     = (float*)carve(NH * 4);
    float*  hw    = (float*)carve(NH4 * 4);          // all 4 relations, fused GEMM
    float*  gi    = (float*)carve(NH3 * 4);
    float*  gh    = (float*)carve(NH3 * 4);
    bf16_t* h_bf  = (bf16_t*)carve(NH * 2);
    bf16_t* m_bf  = (bf16_t*)carve(NH * 2);
    bf16_t* relw_bf = (bf16_t*)carve((size_t)H4 * HDIM * 2);
    bf16_t* wih_bf  = (bf16_t*)carve((size_t)H3 * HDIM * 2);
    bf16_t* whh_bf  = (bf16_t*)carve((size_t)H3 * HDIM * 2);
    float*  c1    = (float*)carve((size_t)64 * L_IN * 4);
    float*  p1    = (float*)carve((size_t)64 * L_P1 * 4);
    float*  c2    = (float*)carve((size_t)64 * L_P1 * 4);
    float*  p2    = (float*)carve((size_t)64 * L_P2 * 4);
    float*  z128  = (float*)carve(128 * 4);
    (void)ws_size;

    const int TB = 256;

    // one-time weight conversion to bf16 (matrix-pipe operand precision)
    k_f32_to_bf16<<<cdiv_u((long)H4 * HDIM, TB), TB, 0, stream>>>(rel_w, relw_bf, (long)H4 * HDIM);
    k_f32_to_bf16<<<cdiv_u((long)H3 * HDIM, TB), TB, 0, stream>>>(gru_w_ih, wih_bf, (long)H3 * HDIM);
    k_f32_to_bf16<<<cdiv_u((long)H3 * HDIM, TB), TB, 0, stream>>>(gru_w_hh, whh_bf, (long)H3 * HDIM);

    // input projection -> x0, h, h_bf16
    k_in_proj<<<cdiv_u((long)NH, TB), TB, 0, stream>>>(x_code, type_ids, type_emb,
                                                       in_proj_w, in_proj_b, x0, h, h_bf);

    // GGNN time steps
    const unsigned gemmH4_blocks = cdiv_u((N_NODES / 16) * (H4 / 64), 8);   // hw_all strips
    const unsigned gemmH3_blocks = cdiv_u((N_NODES / 16) * (H3 / 64), 8);   // gi/gh strips
    for (int step = 0; step < T_STEPS; ++step) {
        k_zero_f32<<<cdiv_u((long)NH, TB), TB, 0, stream>>>(m, (long)NH);
        // hw_all[n, r*128+c] = sum_k h[n,k] * rel_w[r,c,k]  (one fused GEMM)
        k_wmma_gemm_bf16<<<gemmH4_blocks, TB, 0, stream>>>(
            h_bf, relw_bf, nullptr, hw, N_NODES, H4, HDIM);
        for (int r = 0; r < RELN; ++r) {
            k_scatter_add<<<cdiv_u((long)E * 32, TB), TB, 0, stream>>>(
                edges[r], edges[r] + E, hw + (size_t)r * HDIM,
                rel_b + (size_t)r * HDIM, m, E);
        }
        k_f32_to_bf16<<<cdiv_u((long)NH, TB), TB, 0, stream>>>(m, m_bf, (long)NH);
        k_wmma_gemm_bf16<<<gemmH3_blocks, TB, 0, stream>>>(m_bf, wih_bf, gru_b_ih, gi, N_NODES, H3, HDIM);
        k_wmma_gemm_bf16<<<gemmH3_blocks, TB, 0, stream>>>(h_bf, whh_bf, gru_b_hh, gh, N_NODES, H3, HDIM);
        k_gru_gate<<<cdiv_u((long)NH, TB), TB, 0, stream>>>(gi, gh, h, h_bf);
    }

    // conv / pool / fc head
    k_conv1<<<cdiv_u((long)64 * L_IN, TB), TB, 0, stream>>>(x0, h, conv1_w, conv1_b, c1);
    k_pool1<<<cdiv_u((long)64 * L_P1, TB), TB, 0, stream>>>(c1, p1);
    k_conv2<<<cdiv_u((long)64 * L_P1, TB), TB, 0, stream>>>(p1, conv2_w, conv2_b, c2);
    k_pool2<<<cdiv_u((long)64 * L_P2, TB), TB, 0, stream>>>(c2, p2);
    k_fc1<<<128, TB, 0, stream>>>(fc1_w, fc1_b, p2, z128);
    k_fc2<<<1, 64, 0, stream>>>(fc2_w, fc2_b, z128, logits);
}
